// ConditionalRandomField_12309376270659
// MI455X (gfx1250) — matrix operand order repaired and checked by reading, not compiled, not run on hardware
//
#include <hip/hip_runtime.h>
#include <math.h>

typedef __attribute__((ext_vector_type(16))) _Float16 v16h;
typedef __attribute__((ext_vector_type(8)))  _Float16 v8h;
typedef __attribute__((ext_vector_type(8)))  float    v8f;

#define B_  512
#define L_  1024
#define T_  50
#define TP  64                     // T padded to WMMA K/N multiple
#define NEG_BIG (-1.0e30f)         // finite -inf substitute (keeps blend NaN-free)
#define LOG2E 1.4426950408889634f
#define LN2   0.6931471805599453f

// ds_swizzle group-of-32 XOR patterns (and_mask=0x1f; xor<16 keeps 16-lane halves separate)
template<int IMM>
__device__ __forceinline__ float swz_max(float x) {
  int yi = __builtin_amdgcn_ds_swizzle(__builtin_bit_cast(int, x), IMM);
  return fmaxf(x, __builtin_bit_cast(float, yi));
}
template<int IMM>
__device__ __forceinline__ float swz_add(float x) {
  int yi = __builtin_amdgcn_ds_swizzle(__builtin_bit_cast(int, x), IMM);
  return x + __builtin_bit_cast(float, yi);
}
#define SWZ_X1 0x041F
#define SWZ_X2 0x081F
#define SWZ_X4 0x101F
#define SWZ_X8 0x201F

// One wave per 16-row batch tile. Exp-domain forward recurrence with f16 WMMA.
__global__ __launch_bounds__(32)
void crf_forward_kernel(const float* __restrict__ feats,
                        const int*   __restrict__ masks,
                        const float* __restrict__ trans,
                        const float* __restrict__ start,
                        const float* __restrict__ endv,
                        float*       __restrict__ allpath)
{
  __shared__ _Float16 atile[16 * TP];   // 16 batch rows x 64 padded tags, f16

  const int lane  = threadIdx.x;        // 0..31
  const int col15 = lane & 15;
  const int hi    = lane >> 4;          // half-wave select
  const int b0    = blockIdx.x * 16;    // batch tile base

  // ---- Build B fragments once: expT = exp(transition), zero-padded. ----
  // B layout used here (contiguous K, same order as A): VGPR v holds K = kbase+2v, kbase+2v+1; N = lane%16;
  // lanes 0-15 kbase+=0, lanes 16-31 kbase+=16.
  v16h Bf[4][2];
#pragma unroll
  for (int nt = 0; nt < 4; ++nt) {
    const int n = nt * 16 + col15;
#pragma unroll
    for (int kc = 0; kc < 2; ++kc) {
      const int kbase = kc * 32 + (hi ? 16 : 0);
      v16h f;
#pragma unroll
      for (int v = 0; v < 8; ++v) {
        const int k0 = kbase + 2 * v;
        float e0 = (k0     < T_ && n < T_) ? __expf(trans[k0 * T_ + n])       : 0.f;
        float e1 = (k0 + 1 < T_ && n < T_) ? __expf(trans[(k0 + 1) * T_ + n]) : 0.f;
        f[2 * v]     = (_Float16)e0;
        f[2 * v + 1] = (_Float16)e1;
      }
      Bf[nt][kc] = f;
    }
  }

  // ---- alpha0 = start + feats[:,0,:] in C-fragment layout ----
  // C layout: VGPR v, lanes 0-15: (M=v, N=lane); lanes 16-31: (M=8+v, N=lane-16).
  v8f acc[4];
#pragma unroll
  for (int nt = 0; nt < 4; ++nt) {
    const int n = nt * 16 + col15;
#pragma unroll
    for (int v = 0; v < 8; ++v) {
      const int row = v + 8 * hi;
      acc[nt][v] = (n < T_)
                 ? (start[n] + feats[(size_t)(b0 + row) * L_ * T_ + n])
                 : NEG_BIG;
    }
  }

  // ---- Serial forward recurrence over time (branch-free body) ----
  for (int t = 1; t < L_; ++t) {
    // 1) issue all global loads for this step up front (overlap with reduction work)
    float emit[4][8];
    float mk[8];
#pragma unroll
    for (int v = 0; v < 8; ++v) {
      const int row = v + 8 * hi;
      mk[v] = (float)masks[(size_t)(b0 + row) * L_ + t];
      const float* fr = feats + (size_t)(b0 + row) * L_ * T_ + (size_t)t * T_;
#pragma unroll
      for (int nt = 0; nt < 4; ++nt) {
        const int n  = nt * 16 + col15;
        const int nc = (n < T_) ? n : (T_ - 1);     // clamped unconditional load
        float e = fr[nc];
        emit[nt][v] = (n < T_) ? e : 0.f;           // v_cndmask, no branch
      }
    }

    // 2) per-row max of alpha (reduce over 64 source tags: 4 tiles + 16-lane butterfly)
    float m[8];
#pragma unroll
    for (int v = 0; v < 8; ++v) {
      float mv = fmaxf(fmaxf(acc[0][v], acc[1][v]), fmaxf(acc[2][v], acc[3][v]));
      mv = swz_max<SWZ_X1>(mv);
      mv = swz_max<SWZ_X2>(mv);
      mv = swz_max<SWZ_X4>(mv);
      mv = swz_max<SWZ_X8>(mv);
      m[v] = mv;
    }

    // 3) a = exp2((alpha - m)*log2e) -> f16 LDS tile (raw v_exp_f32)
#pragma unroll
    for (int v = 0; v < 8; ++v) {
      const int row = v + 8 * hi;
      const float nm2 = -m[v] * LOG2E;
#pragma unroll
      for (int nt = 0; nt < 4; ++nt) {
        const int n = nt * 16 + col15;
        float a = __builtin_amdgcn_exp2f(fmaf(acc[nt][v], LOG2E, nm2));
        atile[row * TP + n] = (_Float16)a;
      }
    }
    asm volatile("s_wait_dscnt 0" ::: "memory");    // single wave: no barrier needed

    // 4) gather A fragments (16-bit A 16x32 layout) from the LDS tile
    const v8h* base = reinterpret_cast<const v8h*>(&atile[col15 * TP]);
    v8h a0lo = base[0 + 2 * hi];   // K  0.. 7 (lanes<16) / 16..23 (lanes>=16)
    v8h a0hi = base[1 + 2 * hi];   // K  8..15            / 24..31
    v8h a1lo = base[4 + 2 * hi];   // K 32..39            / 48..55
    v8h a1hi = base[5 + 2 * hi];   // K 40..47            / 56..63
    v16h A0, A1;
#pragma unroll
    for (int i = 0; i < 8; ++i) {
      A0[i] = a0lo[i]; A0[8 + i] = a0hi[i];
      A1[i] = a1lo[i]; A1[8 + i] = a1hi[i];
    }

    // 5) exp-domain GEMM + log + emit; arithmetic mask blend (no exec divergence)
#pragma unroll
    for (int nt = 0; nt < 4; ++nt) {
      v8f c = {};
      c = __builtin_amdgcn_wmma_f32_16x16x32_f16(false, A0, false, Bf[nt][0],
                                                 (short)0, c, false, false);
      c = __builtin_amdgcn_wmma_f32_16x16x32_f16(false, A1, false, Bf[nt][1],
                                                 (short)0, c, false, false);
#pragma unroll
      for (int v = 0; v < 8; ++v) {
        float lc = __builtin_amdgcn_logf(c[v]);              // raw v_log_f32 (log2)
        float na = emit[nt][v] + fmaf(lc, LN2, m[v]);        // ln via single fma
        na = fmaxf(na, NEG_BIG);                             // finite: log2(0)=-inf clamped
        acc[nt][v] = fmaf(mk[v], na - acc[nt][v], acc[nt][v]); // mask blend, NaN-free
      }
    }
  }

  // ---- final logsumexp(alpha + end) per batch row ----
#pragma unroll
  for (int v = 0; v < 8; ++v) {
    float vals[4];
    float mv = NEG_BIG;
#pragma unroll
    for (int nt = 0; nt < 4; ++nt) {
      const int n = nt * 16 + col15;
      vals[nt] = (n < T_) ? (acc[nt][v] + endv[n]) : NEG_BIG;
      mv = fmaxf(mv, vals[nt]);
    }
    mv = swz_max<SWZ_X1>(mv);
    mv = swz_max<SWZ_X2>(mv);
    mv = swz_max<SWZ_X4>(mv);
    mv = swz_max<SWZ_X8>(mv);
    float sv = 0.f;
#pragma unroll
    for (int nt = 0; nt < 4; ++nt)
      sv += __builtin_amdgcn_exp2f((vals[nt] - mv) * LOG2E); // padded -> exp2(-huge)=0
    sv = swz_add<SWZ_X1>(sv);
    sv = swz_add<SWZ_X2>(sv);
    sv = swz_add<SWZ_X4>(sv);
    sv = swz_add<SWZ_X8>(sv);
    if (col15 == 0)
      allpath[b0 + v + 8 * hi] = fmaf(__builtin_amdgcn_logf(sv), LN2, mv);
  }
}

// Gold path score + final subtraction. One thread per batch element; memory-bound noise.
__global__ __launch_bounds__(256)
void crf_gold_kernel(const float* __restrict__ feats,
                     const long long* __restrict__ tags,
                     const int*   __restrict__ masks,
                     const float* __restrict__ trans,
                     const float* __restrict__ start,
                     const float* __restrict__ endv,
                     const float* __restrict__ allpath,
                     float*       __restrict__ out)
{
  int b = blockIdx.x * blockDim.x + threadIdx.x;
  if (b >= B_) return;
  const long long* tg   = tags  + (size_t)b * L_;
  const int*       mrow = masks + (size_t)b * L_;
  const float*     frow = feats + (size_t)b * L_ * T_;

  int prev = (int)tg[0];
  float score = start[prev] + frow[prev];
  int cnt = mrow[0];
  for (int t = 1; t < L_; ++t) {
    int   cur = (int)tg[t];
    float mf  = (float)mrow[t];
    score += (frow[(size_t)t * T_ + cur] + trans[prev * T_ + cur]) * mf;
    cnt  += mrow[t];
    prev  = cur;
  }
  int ltag = (int)tg[cnt - 1];
  score += endv[ltag];
  out[b] = allpath[b] - score;
}

extern "C" void kernel_launch(void* const* d_in, const int* in_sizes, int n_in,
                              void* d_out, int out_size, void* d_ws, size_t ws_size,
                              hipStream_t stream) {
  const float*     feats = (const float*)d_in[0];
  const long long* tags  = (const long long*)d_in[1];
  const int*       masks = (const int*)d_in[2];
  const float*     trans = (const float*)d_in[3];
  const float*     start = (const float*)d_in[4];
  const float*     endv  = (const float*)d_in[5];
  float* out     = (float*)d_out;
  float* allpath = (float*)d_ws;           // 512 floats of scratch

  crf_forward_kernel<<<B_ / 16, 32, 0, stream>>>(feats, masks, trans, start, endv, allpath);
  crf_gold_kernel<<<(B_ + 255) / 256, 256, 0, stream>>>(feats, tags, masks, trans,
                                                        start, endv, allpath, out);
}
